// SimpleConditionPolicy_82532091560252
// MI455X (gfx1250) — matrix operand order repaired and checked by reading, not compiled, not run on hardware
//
#include <hip/hip_runtime.h>
#include <math.h>

typedef __attribute__((ext_vector_type(16))) _Float16 v16h;
typedef __attribute__((ext_vector_type(8)))  _Float16 v8h;
typedef __attribute__((ext_vector_type(8)))  float    v8f;

#define ACT     6
#define HID     64
#define GIN     134
#define NSEQ    640
#define TT      128
#define HSTRIDE 96    // h16 row: 64 h | 6 onehot | pad to 96 (192 B, 16B-aligned)

// ---------------------------------------------------------------------------
// Fast activations: one v_exp_f32 + one v_rcp_f32 each (no IEEE divide chain).
// ---------------------------------------------------------------------------
__device__ __forceinline__ float fast_sigmoid(float x)
{
  return __builtin_amdgcn_rcpf(1.f + __expf(-x));
}
__device__ __forceinline__ float fast_tanh(float x)
{
  return 1.f - 2.f * __builtin_amdgcn_rcpf(1.f + __expf(2.f * x));
}

// ---------------------------------------------------------------------------
// Kernel 1: state encoder on frame 0 only.  conv3x3(8->32) -> conv3x3 -> fc.
// ---------------------------------------------------------------------------
__global__ __launch_bounds__(256)
void enc_kernel(const float* __restrict__ s_h,
                const float* __restrict__ c1w, const float* __restrict__ c1b,
                const float* __restrict__ c2w, const float* __restrict__ c2b,
                const float* __restrict__ fcw, const float* __restrict__ fcb,
                float* __restrict__ semb)
{
  __shared__ float s0[512];
  __shared__ float o1[32 * 36];
  __shared__ float o2[512];
  const int seq = blockIdx.x;
  const int tid = threadIdx.x;
  const float* src = s_h + (size_t)seq * (TT * 512);
  for (int i = tid; i < 512; i += 256) s0[i] = src[i];
  __syncthreads();
  for (int idx = tid; idx < 32 * 36; idx += 256) {
    int oc = idx / 36, p = idx % 36, y = p / 6, x = p % 6;
    float acc = c1b[oc];
    for (int c = 0; c < 8; ++c)
      for (int ky = 0; ky < 3; ++ky)
        for (int kx = 0; kx < 3; ++kx)
          acc += s0[c * 64 + (y + ky) * 8 + (x + kx)] *
                 c1w[((oc * 8 + c) * 3 + ky) * 3 + kx];
    o1[idx] = fmaxf(acc, 0.f);
  }
  __syncthreads();
  for (int idx = tid; idx < 512; idx += 256) {
    int oc = idx / 16, p = idx % 16, y = p / 4, x = p % 4;
    float acc = c2b[oc];
    for (int c = 0; c < 32; ++c)
      for (int ky = 0; ky < 3; ++ky)
        for (int kx = 0; kx < 3; ++kx)
          acc += o1[c * 36 + (y + ky) * 6 + (x + kx)] *
                 c2w[((oc * 32 + c) * 3 + ky) * 3 + kx];
    o2[idx] = fmaxf(acc, 0.f);
  }
  __syncthreads();
  if (tid < HID) {
    float acc = fcb[tid];
    for (int k = 0; k < 512; ++k) acc += o2[k] * fcw[tid * 512 + k];
    semb[seq * HID + tid] = fmaxf(acc, 0.f);
  }
}

// ---------------------------------------------------------------------------
// Kernel 2: gx_const (biases folded: b_ih all rows, b_hh only r<128).
// ---------------------------------------------------------------------------
__global__ __launch_bounds__(192)
void gx_kernel(const float* __restrict__ semb, const float* __restrict__ bz,
               const float* __restrict__ wih,  const float* __restrict__ bih,
               const float* __restrict__ bhh,  float* __restrict__ gxc)
{
  __shared__ float se[HID], zz[HID];
  const int seq = blockIdx.x, r = threadIdx.x;
  if (r < HID) { se[r] = semb[seq * HID + r]; zz[r] = bz[(seq / 10) * HID + r]; }
  __syncthreads();
  float acc = bih[r] + (r < 128 ? bhh[r] : 0.f);
  const float* wr = wih + r * GIN;
  for (int k = 0; k < HID; ++k) acc += se[k] * wr[k];
  for (int k = 0; k < HID; ++k) acc += zz[k] * wr[70 + k];
  gxc[seq * 192 + r] = acc;
}

// ---------------------------------------------------------------------------
// Kernel 3: action masks  (a != A-1)
// ---------------------------------------------------------------------------
__global__ void mask_kernel(const int* __restrict__ a_h, float* __restrict__ m, int n)
{
  int i = blockIdx.x * blockDim.x + threadIdx.x;
  if (i < n) m[i] = (a_h[i] != ACT - 1) ? 1.f : 0.f;
}

// ---------------------------------------------------------------------------
// A-fragment loader (ISA 7.12.2, 16-bit A 16x32): lane holds row M = L%16;
// elems 0..7 = K = kbase..kbase+7, elems 8..15 = K = kbase+16..kbase+23.
// Two ds_load_b128 per fragment (all offsets 16B-aligned).
// ---------------------------------------------------------------------------
__device__ __forceinline__ v16h loadA(const _Float16* __restrict__ row, int kbase)
{
  v8h lo = *(const v8h*)(row + kbase);
  v8h hi = *(const v8h*)(row + kbase + 16);
  v16h a;
#pragma unroll
  for (int j = 0; j < 8; ++j) { a[j] = lo[j]; a[8 + j] = hi[j]; }
  return a;
}

// ---------------------------------------------------------------------------
// Kernel 4: persistent GRU rollout, 16 seqs/block, 12 waves.
// Gate input extended to K=96: [ h(64) | onehot(act)(6) | zeros ], with
// tab = emb @ Wih_a.T appended as B-rows 64..69 -> r/z gates are 3 chained
// WMMAs with a constant register seed (no per-step LDS gathers).
// ---------------------------------------------------------------------------
__global__ __launch_bounds__(384)
void recur_kernel(const float* __restrict__ gxc,   // [640][192]
                  const float* __restrict__ whh,   // [192][64]
                  const float* __restrict__ bhh,   // [192]
                  const float* __restrict__ emb,   // [6][6]
                  const float* __restrict__ wih,   // [192][134]
                  const float* __restrict__ m1w,   // [64][64]
                  const float* __restrict__ m1b,   // [64]
                  const float* __restrict__ m2w,   // [6][64]
                  const float* __restrict__ m2b,   // [6]
                  float* __restrict__ out)         // [640][127][6]
{
  __shared__ float    tab[ACT * 192];       // emb @ Wih_a.T (6 actions x 192)
  __shared__ float    rz[2 * 16 * 64];      // [0]=r, [1]=z
  __shared__ float    h32[16 * 64];         // f32 master hidden state
  __shared__ _Float16 h16[16 * HSTRIDE];    // h | onehot | pad
  __shared__ _Float16 hid16[16 * 72];
  __shared__ float    lg[16 * ACT];
  __shared__ int      actS[16];

  const int tid  = threadIdx.x;
  const int w    = tid >> 5, lane = tid & 31;
  const int m16  = lane & 15, half = lane >> 4;
  const int seq0 = blockIdx.x * 16;

  for (int idx = tid; idx < ACT * 192; idx += 384) {
    int a = idx / 192, r = idx % 192;
    float acc = 0.f;
    for (int j = 0; j < ACT; ++j) acc += emb[a * ACT + j] * wih[r * GIN + 64 + j];
    tab[idx] = acc;
  }
  for (int i = tid; i < 16 * 64; i += 384) h32[i] = 0.f;
  for (int i = tid; i < 16 * HSTRIDE; i += 384) h16[i] = (_Float16)0.f;
  for (int i = tid; i < 16 * 72; i += 384) hid16[i] = (_Float16)0.f;
  if (tid < 16) actS[tid] = ACT - 1;
  __syncthreads();
  if (tid < 16) h16[tid * HSTRIDE + 64 + (ACT - 1)] = (_Float16)1.f;  // act0 = 5

  // ---- register-resident weight fragments (loaded once) ----
  v16h bw0 = {}, bw1 = {}, bw2 = {};   // Whh.T cols [w*16,+16): K 0..31 / 32..63 / 64..95
  {
    const int col0 = w * 16;
#pragma unroll
    for (int j = 0; j < 16; ++j) {
      bw0[j] = (_Float16)whh[(col0 + m16) * 64 + (j & 7) + 8 * half + 16 * (j >> 3)];
      bw1[j] = (_Float16)whh[(col0 + m16) * 64 + 32 + (j & 7) + 8 * half + 16 * (j >> 3)];
    }
    if (w < 8) {                       // onehot rows: B row K=64+k = tab[k]
#pragma unroll
      for (int j = 0; j < 16; ++j) {
        const int k = (j & 7) + 8 * half + 16 * (j >> 3);
        bw2[j] = (_Float16)((k < ACT) ? tab[k * 192 + col0 + m16] : 0.f);
      }
    }
  }
  v16h bm10 = {}, bm11 = {};           // mlp1 (waves 0..3)
  if (w < 4) {
    const int col0 = w * 16;
#pragma unroll
    for (int j = 0; j < 16; ++j) {
      bm10[j] = (_Float16)m1w[(col0 + m16) * 64 + (j & 7) + 8 * half + 16 * (j >> 3)];
      bm11[j] = (_Float16)m1w[(col0 + m16) * 64 + 32 + (j & 7) + 8 * half + 16 * (j >> 3)];
    }
  }
  v16h bm20 = {}, bm21 = {};           // mlp2 (wave 0), cols 6..15 zero
  if (w == 0) {
#pragma unroll
    for (int j = 0; j < 16; ++j) {
      bm20[j] = (_Float16)((m16 < ACT) ? m2w[m16 * 64 + (j & 7) + 8 * half + 16 * (j >> 3)] : 0.f);
      bm21[j] = (_Float16)((m16 < ACT) ? m2w[m16 * 64 + 32 + (j & 7) + 8 * half + 16 * (j >> 3)] : 0.f);
    }
  }
  float gf[8];                         // gx_const fragment (N=m16, M=j+8*half)
#pragma unroll
  for (int j = 0; j < 8; ++j)
    gf[j] = gxc[(size_t)(seq0 + j + 8 * half) * 192 + w * 16 + m16];
  const float bhn  = (w >= 8) ? bhh[128 + (w - 8) * 16 + m16] : 0.f;
  const float m1bv = (w < 4)  ? m1b[w * 16 + m16] : 0.f;
  const float m2bv = (w == 0 && m16 < ACT) ? m2b[m16] : 0.f;

  v8f cseed;                           // constant gate seed
#pragma unroll
  for (int j = 0; j < 8; ++j) cseed[j] = (w < 8) ? gf[j] : bhn;

  const int rzBase = (w >> 2) * 1024 + (w & 3) * 16 + m16;
  __syncthreads();

#pragma unroll 1
  for (int t = 0; t < TT - 1; ++t) {
    // ---- gates: gh(+gx via onehot rows) ----
    const _Float16* hrow = &h16[m16 * HSTRIDE];
    v16h a0 = loadA(hrow, half * 8);
    v16h a1 = loadA(hrow, half * 8 + 32);
    v8f c = cseed;
    c = __builtin_amdgcn_wmma_f32_16x16x32_f16(false, a0, false, bw0, (short)0, c, false, false);
    c = __builtin_amdgcn_wmma_f32_16x16x32_f16(false, a1, false, bw1, (short)0, c, false, false);
    if (w < 8) {                       // + onehot @ tab  (r/z gates only)
      v16h a2 = loadA(hrow, half * 8 + 64);
      c = __builtin_amdgcn_wmma_f32_16x16x32_f16(false, a2, false, bw2, (short)0, c, false, false);
#pragma unroll
      for (int j = 0; j < 8; ++j)
        rz[rzBase + (j + 8 * half) * 64] = fast_sigmoid(c[j]);
    }
    __syncthreads();                   // B1

    if (w >= 8) {                      // n gate + h update (f32 master state)
#pragma unroll
      for (int j = 0; j < 8; ++j) {
        int m = j + 8 * half;
        int col = (w - 8) * 16 + m16;
        float gxn  = gf[j] + tab[actS[m] * 192 + 128 + col];
        float n    = fast_tanh(gxn + rz[m * 64 + col] * c[j]);
        float z    = rz[1024 + m * 64 + col];
        float hnew = (1.f - z) * n + z * h32[m * 64 + col];
        h32[m * 64 + col] = hnew;
        h16[m * HSTRIDE + col] = (_Float16)hnew;
      }
    }
    __syncthreads();                   // B2

    if (w < 4) {                       // mlp1: tanh(h_new @ W1.T + b1)
      v16h p0 = loadA(&h16[m16 * HSTRIDE], half * 8);
      v16h p1 = loadA(&h16[m16 * HSTRIDE], half * 8 + 32);
      v8f cm;
#pragma unroll
      for (int j = 0; j < 8; ++j) cm[j] = m1bv;
      cm = __builtin_amdgcn_wmma_f32_16x16x32_f16(false, p0, false, bm10, (short)0, cm, false, false);
      cm = __builtin_amdgcn_wmma_f32_16x16x32_f16(false, p1, false, bm11, (short)0, cm, false, false);
#pragma unroll
      for (int j = 0; j < 8; ++j) {
        int m = j + 8 * half;
        hid16[m * 72 + w * 16 + m16] = (_Float16)fast_tanh(cm[j]);
      }
    }
    __syncthreads();                   // B3

    if (w == 0) {                      // mlp2 -> logits -> global; argmax; onehot
      v16h p0 = loadA(&hid16[m16 * 72], half * 8);
      v16h p1 = loadA(&hid16[m16 * 72], half * 8 + 32);
      v8f cm;
#pragma unroll
      for (int j = 0; j < 8; ++j) cm[j] = m2bv;
      cm = __builtin_amdgcn_wmma_f32_16x16x32_f16(false, p0, false, bm20, (short)0, cm, false, false);
      cm = __builtin_amdgcn_wmma_f32_16x16x32_f16(false, p1, false, bm21, (short)0, cm, false, false);
      if (m16 < ACT) {
#pragma unroll
        for (int j = 0; j < 8; ++j) {
          int m = j + 8 * half;
          out[(size_t)(seq0 + m) * (TT - 1) * ACT + (size_t)t * ACT + m16] = cm[j];
          lg[m * ACT + m16] = cm[j];
        }
      }
      // same-wave LDS ordering: ds ops from one wave are processed in order,
      // so lanes 0..15 can read lg written above without a barrier.
      if (lane < 16) {
        float best = lg[lane * ACT]; int bi = 0;
#pragma unroll
        for (int a = 1; a < ACT; ++a) {
          float v = lg[lane * ACT + a];
          if (v > best) { best = v; bi = a; }
        }
        actS[lane] = bi;
        // one-hot f16 row (3 packed dword stores): 1.0h = 0x3C00
        unsigned p[3];
#pragma unroll
        for (int q = 0; q < 3; ++q)
          p[q] = (bi == 2 * q ? 0x3C00u : 0u) | (bi == 2 * q + 1 ? 0x3C000000u : 0u);
        unsigned* oh = (unsigned*)&h16[lane * HSTRIDE + 64];
        oh[0] = p[0]; oh[1] = p[1]; oh[2] = p[2];
      }
    }
    __syncthreads();                   // B4
  }
}

// ---------------------------------------------------------------------------
extern "C" void kernel_launch(void* const* d_in, const int* in_sizes, int n_in,
                              void* d_out, int out_size, void* d_ws, size_t ws_size,
                              hipStream_t stream)
{
  const float* s_h = (const float*)d_in[0];
  const int*   a_h = (const int*)d_in[1];
  const float* b_z = (const float*)d_in[2];
  const float* c1w = (const float*)d_in[3];
  const float* c1b = (const float*)d_in[4];
  const float* c2w = (const float*)d_in[5];
  const float* c2b = (const float*)d_in[6];
  const float* fcw = (const float*)d_in[7];
  const float* fcb = (const float*)d_in[8];
  const float* emb = (const float*)d_in[9];
  const float* wih = (const float*)d_in[10];
  const float* whh = (const float*)d_in[11];
  const float* bih = (const float*)d_in[12];
  const float* bhh = (const float*)d_in[13];
  const float* m1w = (const float*)d_in[14];
  const float* m1b = (const float*)d_in[15];
  const float* m2w = (const float*)d_in[16];
  const float* m2b = (const float*)d_in[17];

  float* out_logits = (float*)d_out;                              // 640*127*6
  float* out_mask   = out_logits + (size_t)NSEQ * (TT - 1) * ACT; // 640*128

  float* ws_gx   = (float*)d_ws;              // 640*192 f32
  float* ws_semb = ws_gx + NSEQ * 192;        // 640*64  f32

  hipLaunchKernelGGL(enc_kernel, dim3(NSEQ), dim3(256), 0, stream,
                     s_h, c1w, c1b, c2w, c2b, fcw, fcb, ws_semb);
  hipLaunchKernelGGL(gx_kernel, dim3(NSEQ), dim3(192), 0, stream,
                     ws_semb, b_z, wih, bih, bhh, ws_gx);
  hipLaunchKernelGGL(mask_kernel, dim3((NSEQ * TT + 255) / 256), dim3(256), 0, stream,
                     a_h, out_mask, NSEQ * TT);
  hipLaunchKernelGGL(recur_kernel, dim3(NSEQ / 16), dim3(384), 0, stream,
                     ws_gx, whh, bhh, emb, wih, m1w, m1b, m2w, m2b, out_logits);
}
